// MultiHeadAttention_14362370637847
// MI455X (gfx1250) — matrix operand order repaired and checked by reading, not compiled, not run on hardware
//
#include <hip/hip_runtime.h>
#include <stdint.h>

typedef __attribute__((ext_vector_type(16))) __bf16   v16bf;
typedef __attribute__((ext_vector_type(8)))  float    v8f;
typedef __attribute__((ext_vector_type(4)))  uint32_t u32x4;
typedef __attribute__((ext_vector_type(4)))  int      v4i;
typedef __attribute__((ext_vector_type(8)))  int      i32x8;
typedef __attribute__((ext_vector_type(4)))  int      i32x4;

union Frag { v16bf v; uint32_t u[8]; };
union Acc  { v8f v; float f[8]; };
union U4x  { uint4 q; __bf16 b[8]; };

static constexpr int NH   = 12;        // heads
static constexpr int SEQ  = 128;       // T
static constexpr int DM   = 768;       // d_model
static constexpr int HDIM = 64;        // head dim
static constexpr int QKVS = 3 * DM;    // fused QKV row stride (2304)

// ---------------------------------------------------------------------------
// CDNA5 async global->LDS copy (ASYNCcnt), guarded with sync fallback
// ---------------------------------------------------------------------------
#if defined(__gfx1250__) && __has_builtin(__builtin_amdgcn_global_load_async_to_lds_b128)
#define USE_ASYNC_COPY 1
#else
#define USE_ASYNC_COPY 0
#endif

#if USE_ASYNC_COPY
#if __has_builtin(__builtin_amdgcn_s_wait_asynccnt)
#define WAIT_ASYNC() __builtin_amdgcn_s_wait_asynccnt(0)
#else
#define WAIT_ASYNC() asm volatile("s_wait_asynccnt 0x0" ::: "memory")
#endif
typedef __attribute__((address_space(1))) v4i* gv4i_p;  // global <4 x i32>*
typedef __attribute__((address_space(3))) v4i* lv4i_p;  // LDS    <4 x i32>*
#else
#define WAIT_ASYNC() ((void)0)
#endif

// ---------------------------------------------------------------------------
// CDNA5 Tensor Data Mover (TENSORcnt), guarded; arity differs per toolchain
// ---------------------------------------------------------------------------
#if defined(__gfx1250__) && __has_builtin(__builtin_amdgcn_tensor_load_to_lds) && \
    __has_builtin(__builtin_amdgcn_s_wait_tensorcnt)
#define USE_TDM 1
#else
#define USE_TDM 0
#endif

static __device__ __forceinline__ void cp16(const void* g, void* lds) {
#if USE_ASYNC_COPY
  __builtin_amdgcn_global_load_async_to_lds_b128(
      (gv4i_p)(uintptr_t)g,
      (lv4i_p)(uint32_t)(uintptr_t)lds,
      0, 0);
#else
  *(u32x4*)lds = *(const u32x4*)g;
#endif
}

#if USE_TDM
// TDM: load a 128-row x 64-elem bf16 tile (global row stride QKVS elements)
// into LDS rows padded to 144B (128B data + 16B pad) in a single instruction.
// D# per CDNA5 ISA:
//   pad_interval=4 (2^(4+1)=32 dwords = 128B), pad_amount=3 (4 dwords = 16B)
static __device__ __forceinline__ void tdm_load_tile_128x64(const __bf16* gbase,
                                                            uint32_t lds_off) {
  const uint64_t ga = (uint64_t)(uintptr_t)gbase;
  u32x4 g0;
  g0[0] = 1u;                                   // count=1 (valid user descriptor)
  g0[1] = lds_off;                              // lds_addr (bytes)
  g0[2] = (uint32_t)ga;                         // global_addr[31:0]
  g0[3] = (uint32_t)(ga >> 32) | (2u << 30);    // global_addr[56:32] | type=2
  i32x8 g1;
  g1[0] = (int)((1u << 16)     |                // data_size = 2 bytes
                (1u << 20)     |                // pad_enable
                (4u << 22)     |                // pad_interval: 32 dwords
                (3u << 25));                    // pad_amount: 4 dwords
  g1[1] = (int)(64u << 16);                     // tensor_dim0 = 64
  g1[2] = (int)(128u << 16);                    // tensor_dim1 = 128
  g1[3] = (int)(64u << 16);                     // tile_dim0 = 64
  g1[4] = 128;                                  // tile_dim1 = 128 (tile_dim2=0)
  g1[5] = QKVS;                                 // tensor_dim0_stride = 2304
  g1[6] = 0;
  g1[7] = 0;
  const i32x4 z4 = (i32x4){0, 0, 0, 0};
#if __clang_major__ >= 23
  const i32x8 z8 = (i32x8){0, 0, 0, 0, 0, 0, 0, 0};
  __builtin_amdgcn_tensor_load_to_lds(g0, g1, z4, z4, z8, 0);
#else
  __builtin_amdgcn_tensor_load_to_lds(g0, g1, z4, z4, 0);
#endif
}
#endif

static __device__ __forceinline__ uint32_t pack2(float a, float b) {
  union { __bf16 h[2]; uint32_t u; } p;
  p.h[0] = (__bf16)a; p.h[1] = (__bf16)b;
  return p.u;
}

static __device__ __forceinline__ v8f wmma_bf16(const Frag& a, const Frag& b, v8f c) {
  return __builtin_amdgcn_wmma_f32_16x16x32_bf16(false, a.v, false, b.v, (short)0, c, false, false);
}

// load one 16x16x32 operand as two aligned b128 chunks
static __device__ __forceinline__ void ld_frag(Frag& f, const char* c0p, const char* c1p) {
  u32x4 c0 = *(const u32x4*)c0p;
  u32x4 c1 = *(const u32x4*)c1p;
  f.u[0] = c0[0]; f.u[1] = c0[1]; f.u[2] = c0[2]; f.u[3] = c0[3];
  f.u[4] = c1[0]; f.u[5] = c1[1]; f.u[6] = c1[2]; f.u[7] = c1[3];
}

// ---------------------------------------------------------------------------
// x fp32 -> bf16 (one pass)
// ---------------------------------------------------------------------------
__global__ __launch_bounds__(256) void xconv_kernel(const float* __restrict__ x,
                                                    __bf16* __restrict__ xb, size_t n) {
  size_t i = ((size_t)blockIdx.x * 256 + threadIdx.x) * 8;
  if (i >= n) return;
  float4 f0 = *(const float4*)(x + i);
  float4 f1 = *(const float4*)(x + i + 4);
  uint32_t* d = (uint32_t*)(xb + i);
  d[0] = pack2(f0.x, f0.y); d[1] = pack2(f0.z, f0.w);
  d[2] = pack2(f1.x, f1.y); d[3] = pack2(f1.z, f1.w);
}

// ---------------------------------------------------------------------------
// weights -> bf16, transposed to [N][K]; Wq/Wk/Wv fused into [2304][768]
// ---------------------------------------------------------------------------
__global__ __launch_bounds__(256) void wconv_t_kernel(const float* __restrict__ wq,
                                                      const float* __restrict__ wk,
                                                      const float* __restrict__ wv,
                                                      const float* __restrict__ wo,
                                                      __bf16* __restrict__ wqkv_t,
                                                      __bf16* __restrict__ wo_t) {
  int idx = blockIdx.x * 256 + threadIdx.x;
  if (idx >= DM * DM) return;
  int n = idx / DM, k = idx - n * DM;
  size_t src = (size_t)k * DM + n;
  wqkv_t[(size_t)n * DM + k]              = (__bf16)wq[src];
  wqkv_t[(size_t)(DM + n) * DM + k]       = (__bf16)wk[src];
  wqkv_t[(size_t)(2 * DM + n) * DM + k]   = (__bf16)wv[src];
  wo_t  [(size_t)n * DM + k]              = (__bf16)wo[src];
}

// ---------------------------------------------------------------------------
// WMMA GEMM: C[M,N] = A[M,K] * Bt[N,K]^T (+bias). A,Bt bf16.
// 256 threads = 8 waves; tile 128x128, K-step 32, double-buffered async LDS.
// LDS rows padded to 80B (odd multiple of 16B: aligned b128 + few conflicts).
// ---------------------------------------------------------------------------
template<bool OUT_F32_BIAS>
__global__ __launch_bounds__(256)
void gemm_kernel(const __bf16* __restrict__ A, const __bf16* __restrict__ Bt,
                 void* __restrict__ Cp, const float* __restrict__ bias,
                 int M, int N, int Kd) {
  constexpr int BK = 32;
  constexpr int LSTR = 40;  // elements per LDS row (80 bytes)
  __shared__ alignas(16) __bf16 As[2][128 * LSTR];
  __shared__ alignas(16) __bf16 Bs[2][128 * LSTR];

  const int tid    = threadIdx.x;
  const int lane   = tid & 31;
  const int lanelo = lane & 15;
  const int half   = lane >> 4;
  const int wave   = tid >> 5;
  const int wm     = wave & 3;   // M group of 32
  const int wn     = wave >> 2;  // N group of 64
  const int m0     = blockIdx.y * 128;
  const int n0     = blockIdx.x * 128;

  const int srow = tid >> 1;        // staging row 0..127
  const int scol = (tid & 1) * 16;  // element offset 0/16

  auto stage = [&](int buf, int kb) {
    const __bf16* ga = A  + (size_t)(m0 + srow) * Kd + kb + scol;
    const __bf16* gb = Bt + (size_t)(n0 + srow) * Kd + kb + scol;
    __bf16* la = &As[buf][srow * LSTR + scol];
    __bf16* lb = &Bs[buf][srow * LSTR + scol];
    cp16(ga,     la);     cp16(ga + 8, la + 8);
    cp16(gb,     lb);     cp16(gb + 8, lb + 8);
  };

  Acc acc[2][4];
  #pragma unroll
  for (int mi = 0; mi < 2; mi++)
    #pragma unroll
    for (int ni = 0; ni < 4; ni++)
      acc[mi][ni].v = (v8f){0.f,0.f,0.f,0.f,0.f,0.f,0.f,0.f};

  stage(0, 0);
  const int nk = Kd / BK;
  for (int ki = 0; ki < nk; ki++) {
    const int buf = ki & 1;
    WAIT_ASYNC();
    __syncthreads();
    if (ki + 1 < nk) stage(buf ^ 1, (ki + 1) * BK);

    Frag af[2];
    #pragma unroll
    for (int mi = 0; mi < 2; mi++) {
      // A frag: elems 0-7 at k=half*8, elems 8-15 at k=16+half*8
      const char* base = (const char*)&As[buf][(wm * 32 + mi * 16 + lanelo) * LSTR] + half * 16;
      ld_frag(af[mi], base, base + 32);
    }
    #pragma unroll
    for (int ni = 0; ni < 4; ni++) {
      // B frag: elems 0-7 at k=half*8, elems 8-15 at k=16+half*8
      Frag bf;
      const char* base = (const char*)&Bs[buf][(wn * 64 + ni * 16 + lanelo) * LSTR] + half * 16;
      ld_frag(bf, base, base + 32);
      #pragma unroll
      for (int mi = 0; mi < 2; mi++)
        acc[mi][ni].v = wmma_bf16(af[mi], bf, acc[mi][ni].v);
    }
  }

  #pragma unroll
  for (int mi = 0; mi < 2; mi++)
    #pragma unroll
    for (int ni = 0; ni < 4; ni++) {
      const int col = n0 + wn * 64 + ni * 16 + lanelo;
      #pragma unroll
      for (int r = 0; r < 8; r++) {
        const int row = m0 + wm * 32 + mi * 16 + half * 8 + r;
        if constexpr (OUT_F32_BIAS) {
          ((float*)Cp)[(size_t)row * N + col] = acc[mi][ni].f[r] + bias[col];
        } else {
          ((__bf16*)Cp)[(size_t)row * N + col] = (__bf16)acc[mi][ni].f[r];
        }
      }
    }
}

// ---------------------------------------------------------------------------
// Fused causal attention per (b,h): ctx = softmax(Q K^T / 8) V
// 256 threads = 8 waves; wave w owns 16 rows.
// Q/K staged by TDM (tensor_load_to_lds, TENSORcnt) with LDS row padding,
// falling back to async copies; V transposed in LDS; P in LDS.
// All WMMA fragments = 2x ds_load_b128.
// ---------------------------------------------------------------------------
__global__ __launch_bounds__(256)
void attn_kernel(const __bf16* __restrict__ QKV, __bf16* __restrict__ ctx) {
  constexpr int QSTR = 72;   // Qs/Ks row stride in elements (144B = 128B + 16B pad)
  constexpr int TSTR = 136;  // Vt/Ps row stride in elements (272B)
  extern __shared__ __align__(16) __bf16 smem[];
  __bf16* Qs = smem;                 // 128*72
  __bf16* Ks = Qs + SEQ * QSTR;      // 128*72
  __bf16* Vt = Ks + SEQ * QSTR;      // 64*136  (V transposed: [d][t])
  __bf16* Ps = Vt + HDIM * TSTR;     // 128*136

  const int tid    = threadIdx.x;
  const int lane   = tid & 31;
  const int lanelo = lane & 15;
  const int half   = lane >> 4;
  const int wave   = tid >> 5;   // 0..7
  const int b      = blockIdx.x / NH;
  const int h      = blockIdx.x % NH;
  const int rbase  = wave * 16;

  // ---- stage Q,K (TDM or async), build V^T (scatter) ----
  {
    const int trow = tid >> 1;
    const int seg  = (tid & 1) * 32;  // element offset within the 64-wide head slice
    const size_t rowbase = ((size_t)(b * SEQ + trow)) * QKVS + (size_t)h * HDIM;
#if USE_TDM
    if (wave == 0) {
      const __bf16* g0 = QKV + (size_t)(b * SEQ) * QKVS + (size_t)h * HDIM;
      tdm_load_tile_128x64(g0,      (uint32_t)(uintptr_t)Qs);
      tdm_load_tile_128x64(g0 + DM, (uint32_t)(uintptr_t)Ks);
    }
#else
    const __bf16* gq = QKV + rowbase + seg;            // Q block
    const __bf16* gk = QKV + rowbase + DM + seg;       // K block
    __bf16* lq = Qs + trow * QSTR + seg;
    __bf16* lk = Ks + trow * QSTR + seg;
    #pragma unroll
    for (int j = 0; j < 4; j++) {
      cp16(gq + j * 8, lq + j * 8);
      cp16(gk + j * 8, lk + j * 8);
    }
#endif
    const uint4* gv = (const uint4*)(QKV + rowbase + 2 * DM + seg);
    #pragma unroll
    for (int i = 0; i < 4; i++) {
      U4x u; u.q = gv[i];
      #pragma unroll
      for (int j = 0; j < 8; j++)
        Vt[(seg + i * 8 + j) * TSTR + trow] = u.b[j];
    }
  }
#if USE_TDM
  if (wave == 0) __builtin_amdgcn_s_wait_tensorcnt(0);
#else
  WAIT_ASYNC();
#endif
  __syncthreads();

  // ---- S = Q K^T (16 rows x 128 cols per wave), K-dim 64 = 2 steps ----
  Acc s[8];
  #pragma unroll
  for (int ni = 0; ni < 8; ni++) s[ni].v = (v8f){0.f,0.f,0.f,0.f,0.f,0.f,0.f,0.f};

  #pragma unroll
  for (int ks = 0; ks < 2; ks++) {
    const int d0 = ks * 32;
    Frag aq;
    {
      const char* base = (const char*)(Qs + (rbase + lanelo) * QSTR) + d0 * 2 + half * 16;
      ld_frag(aq, base, base + 32);
    }
    #pragma unroll
    for (int ni = 0; ni < 8; ni++) {
      Frag bk;
      const char* base = (const char*)(Ks + (ni * 16 + lanelo) * QSTR) + d0 * 2 + half * 16;
      ld_frag(bk, base, base + 32);
      s[ni].v = wmma_bf16(aq, bk, s[ni].v);
    }
  }

  // ---- causal mask + softmax per row (row = rbase + 8*half + r) ----
  const float scale = 0.125f;  // 1/sqrt(64)
  #pragma unroll
  for (int r = 0; r < 8; r++) {
    const int row = rbase + half * 8 + r;
    float mx = -3.0e38f;
    #pragma unroll
    for (int ni = 0; ni < 8; ni++) {
      const int col = ni * 16 + lanelo;
      float sv = s[ni].f[r] * scale;
      sv = (col <= row) ? sv : -3.0e38f;
      s[ni].f[r] = sv;
      mx = fmaxf(mx, sv);
    }
    #pragma unroll
    for (int off = 1; off < 16; off <<= 1) mx = fmaxf(mx, __shfl_xor(mx, off, 32));
    float sum = 0.f;
    #pragma unroll
    for (int ni = 0; ni < 8; ni++) {
      float p = __expf(s[ni].f[r] - mx);
      s[ni].f[r] = p;
      sum += p;
    }
    #pragma unroll
    for (int off = 1; off < 16; off <<= 1) sum += __shfl_xor(sum, off, 32);
    const float inv = 1.0f / sum;
    #pragma unroll
    for (int ni = 0; ni < 8; ni++)
      Ps[row * TSTR + ni * 16 + lanelo] = (__bf16)(s[ni].f[r] * inv);
  }
  __syncthreads();

  // ---- ctx = P V (M=16, N=64, K=128 = 4 steps) ----
  Acc o[4];
  #pragma unroll
  for (int ni = 0; ni < 4; ni++) o[ni].v = (v8f){0.f,0.f,0.f,0.f,0.f,0.f,0.f,0.f};

  #pragma unroll
  for (int kt = 0; kt < 4; kt++) {
    const int k0 = kt * 32;
    Frag ap;
    {
      const char* base = (const char*)(Ps + (rbase + lanelo) * TSTR) + k0 * 2 + half * 16;
      ld_frag(ap, base, base + 32);
    }
    #pragma unroll
    for (int ni = 0; ni < 4; ni++) {
      Frag bv;
      const char* base = (const char*)(Vt + (ni * 16 + lanelo) * TSTR) + k0 * 2 + half * 16;
      ld_frag(bv, base, base + 32);
      o[ni].v = wmma_bf16(ap, bv, o[ni].v);
    }
  }

  // ---- store ctx in [B*T, 768] layout for the output projection ----
  #pragma unroll
  for (int ni = 0; ni < 4; ni++) {
    const int col = ni * 16 + lanelo;
    #pragma unroll
    for (int r = 0; r < 8; r++) {
      const int row = rbase + half * 8 + r;
      ctx[((size_t)(b * SEQ + row)) * DM + (size_t)h * HDIM + col] = (__bf16)o[ni].f[r];
    }
  }
}

// ---------------------------------------------------------------------------
extern "C" void kernel_launch(void* const* d_in, const int* in_sizes, int n_in,
                              void* d_out, int out_size, void* d_ws, size_t ws_size,
                              hipStream_t stream) {
  (void)in_sizes; (void)n_in; (void)out_size; (void)ws_size;
  const float* x  = (const float*)d_in[0];
  const float* Wq = (const float*)d_in[1];
  const float* Wk = (const float*)d_in[2];
  const float* Wv = (const float*)d_in[3];
  const float* Wo = (const float*)d_in[4];
  const float* bo = (const float*)d_in[5];
  float* out = (float*)d_out;

  const int B = 512;
  const int M = B * SEQ;  // 65536 tokens

  char* ws = (char*)d_ws;
  __bf16* wqkv_t = (__bf16*)ws;  ws += (size_t)QKVS * DM * sizeof(__bf16);
  __bf16* wo_t   = (__bf16*)ws;  ws += (size_t)DM * DM * sizeof(__bf16);
  __bf16* Xb     = (__bf16*)ws;  ws += (size_t)M * DM * sizeof(__bf16);
  __bf16* QKVb   = (__bf16*)ws;  ws += (size_t)M * QKVS * sizeof(__bf16);
  __bf16* Ctx    = (__bf16*)ws;  ws += (size_t)M * DM * sizeof(__bf16);

  // 1) x -> bf16, weights -> bf16 transposed [N][K]
  const size_t nx = (size_t)M * DM;
  xconv_kernel<<<(int)(nx / 8 / 256), 256, 0, stream>>>(x, Xb, nx);
  wconv_t_kernel<<<(DM * DM + 255) / 256, 256, 0, stream>>>(Wq, Wk, Wv, Wo, wqkv_t, wo_t);

  // 2) fused QKV projection: [M,768] x [768,2304] -> [M,2304]
  dim3 gqkv(QKVS / 128, M / 128);  // (18, 512)
  gemm_kernel<false><<<gqkv, 256, 0, stream>>>(Xb, wqkv_t, QKVb, nullptr, M, QKVS, DM);

  // 3) fused causal attention per (b,h)
  const size_t smem = (size_t)(2 * SEQ * 72 + HDIM * 136 + SEQ * 136) * sizeof(__bf16);
  attn_kernel<<<B * NH, 256, smem, stream>>>(QKVb, Ctx);

  // 4) output projection with bias, fp32 out
  dim3 gout(DM / 128, M / 128);    // (6, 512)
  gemm_kernel<true><<<gout, 256, 0, stream>>>(Ctx, wo_t, out, bo, M, DM, DM);
}
